// MaskedSplineFlows_33414845562909
// MI455X (gfx1250) — compile-verified, matrix operand order checked by reading, not run on hardware
//
#include <hip/hip_runtime.h>
#include <hip/hip_bf16.h>

// ---------------- problem constants ----------------
constexpr int   kBATCH = 16384;
constexpr int   kD     = 64;       // DX == DY
constexpr int   kW     = 128;      // hidden WIDTH
constexpr int   kL     = 8;        // LAYERS
constexpr int   kP     = 23;       // 3*KNOTS-1
constexpr int   kDP    = kD * kP;  // 1472
constexpr float kBI    = 4.0f;     // INTERVAL
constexpr float kADJ   = 0.01f;    // SOFTMAX_ADJ
constexpr float kMD    = 0.001f;   // MIN_DERIV

constexpr int   kM     = 32;       // batch rows per workgroup
constexpr int   kPS    = 1480;     // padded param stride (f32 words)
constexpr int   kNT2   = kDP / 16; // 92 n-tiles in GEMM2

typedef float  v8f    __attribute__((ext_vector_type(8)));
typedef __bf16 bf16x8 __attribute__((ext_vector_type(8)));
typedef __bf16 bf16x16 __attribute__((ext_vector_type(16)));

// ---------------- weight mask+convert kernels ----------------
// m1[h][i] = ((h % (D-1)) + 1 >= i+1)  ->  (h % 63) >= i
__global__ void cvt_w1_kernel(const float* __restrict__ src, __bf16* __restrict__ dst) {
    int idx = blockIdx.x * 256 + threadIdx.x;              // kL*kW*kD = 65536
    int i = idx & (kD - 1);
    int h = (idx >> 6) & (kW - 1);
    bool keep = (h % (kD - 1)) >= i;
    dst[idx] = (__bf16)(keep ? src[idx] : 0.0f);
}

// m2[o][h] = (o/23 + 1 > (h%63)+1) -> (o/23) > (h%63)
__global__ void cvt_wout_kernel(const float* __restrict__ src, __bf16* __restrict__ dst, int n) {
    int idx = blockIdx.x * 256 + threadIdx.x;              // kL*kDP*kW = 1507328
    if (idx >= n) return;
    int h = idx & (kW - 1);
    int o = (idx >> 7) % kDP;
    bool keep = (o / kP) > (h % (kD - 1));
    dst[idx] = (__bf16)(keep ? src[idx] : 0.0f);
}

// ---------------- fragment helpers ----------------
// 16-bit A/B layout (ISA 7.12.2): lane holds row/col = lane%16;
// lanes 0-15: K = {k0..k0+7, k0+16..k0+23}; lanes 16-31: +8 on the base.
__device__ __forceinline__ bf16x16 ld_frag(const __bf16* p) {
    bf16x8 lo = *(const bf16x8*)(p);
    bf16x8 hi = *(const bf16x8*)(p + 16);
    return __builtin_shufflevector(lo, hi, 0,1,2,3,4,5,6,7,8,9,10,11,12,13,14,15);
}

__device__ __forceinline__ v8f wmma_bf16(bf16x16 a, bf16x16 b, v8f c) {
    return __builtin_amdgcn_wmma_f32_16x16x32_bf16(
        /*neg_a=*/false, a, /*neg_b=*/false, b,
        /*c_mod=*/(short)0, c, /*reuse_a=*/false, /*reuse_b=*/false);
}

__device__ __forceinline__ float fast_rcp(float x) {
    return __builtin_amdgcn_rcpf(x);   // v_rcp_f32
}

__device__ __forceinline__ float softplusf(float v) {
    float e = __expf(v);
    float r = __logf(1.0f + e);
    return (v > 20.0f) ? v : r;   // exp overflow guard; v<-20 path is exact enough
}

// adjusted softmax over 8 values, scaled by 2*BI
__device__ __forceinline__ void adj_softmax8(const float* v, float* o) {
    float m = v[0];
#pragma unroll
    for (int j = 1; j < 8; ++j) m = fmaxf(m, v[j]);
    float e[8]; float s = 0.0f;
#pragma unroll
    for (int j = 0; j < 8; ++j) { e[j] = __expf(v[j] - m); s += e[j]; }
    float inv = fast_rcp(s);
    const float post = (2.0f * kBI) / (1.0f + kADJ);
    const float addc = kADJ / 8.0f;
#pragma unroll
    for (int j = 0; j < 8; ++j) o[j] = (e[j] * inv + addc) * post;
}

// ---------------- main fused flow kernel ----------------
__global__ __launch_bounds__(256)
void maf_flow_kernel(const float* __restrict__ xin,
                     const __bf16* __restrict__ W1b,   // [L][128][64] masked bf16
                     const float*  __restrict__ b1,    // [L][128]
                     const __bf16* __restrict__ Woutb, // [L][1472][128] masked bf16
                     const float*  __restrict__ bout,  // [L][1472]
                     float* __restrict__ out_x,
                     float* __restrict__ out_ld)
{
    __shared__ float  Pbuf[kM * kPS];     // 189,440 B  spline params (f32)
    __shared__ float  Xf[2][kM * kD];     //  16,384 B  ping-pong activations
    __shared__ __bf16 Xbf[kM * kD];       //   4,096 B  A operand GEMM1
    __shared__ __bf16 Hbf[kM * kW];       //   8,192 B  A operand GEMM2
    __shared__ float  ldacc[kM];          //     128 B  log-det accumulators

    const int t     = threadIdx.x;
    const int lane  = t & 31;
    const int wave  = t >> 5;
    const int laneN = lane & 15;
    const int koff  = (lane >> 4) * 8;    // K-half base / C-row offset
    const int row0  = (int)blockIdx.x * kM;

    // ---- load batch tile ----
#pragma unroll
    for (int i = 0; i < (kM * kD) / 256; ++i) {
        int idx = i * 256 + t;
        Xf[0][idx] = xin[(size_t)row0 * kD + idx];
    }
    if (t < kM) ldacc[t] = 0.0f;
    __syncthreads();

    int cur = 0;
    for (int layer = 0; layer < kL; ++layer) {
        // ---- stage current x as bf16 for GEMM1 ----
#pragma unroll
        for (int i = 0; i < (kM * kD) / 256; ++i) {
            int idx = i * 256 + t;
            Xbf[idx] = (__bf16)Xf[cur][idx];
        }
        __syncthreads();

        // ---- GEMM1: H[32x128] = relu(X[32x64] * W1m^T + b1) ----
        {
            const __bf16* wl = W1b + (size_t)layer * kW * kD;
            const __bf16* bp = wl + (wave * 16 + laneN) * kD + koff; // col = hidden unit
            bf16x16 bf0 = ld_frag(bp);        // K 0..31
            bf16x16 bf1 = ld_frag(bp + 32);   // K 32..63
            float bias = b1[layer * kW + wave * 16 + laneN];
            int col = wave * 16 + laneN;
#pragma unroll
            for (int m = 0; m < 2; ++m) {
                const __bf16* ap = &Xbf[(m * 16 + laneN) * kD + koff];
                bf16x16 a0 = ld_frag(ap);
                bf16x16 a1 = ld_frag(ap + 32);
                v8f acc = {0.f,0.f,0.f,0.f,0.f,0.f,0.f,0.f};
                acc = wmma_bf16(a0, bf0, acc);
                acc = wmma_bf16(a1, bf1, acc);
                int rbase = m * 16 + koff;
#pragma unroll
                for (int r = 0; r < 8; ++r) {
                    float v = fmaxf(acc[r] + bias, 0.0f);
                    Hbf[(rbase + r) * kW + col] = (__bf16)v;
                }
            }
        }
        __syncthreads();

        // ---- GEMM2: P[32x1472] = H[32x128] * Woutm^T + bout ----
        // Software-pipelined over n-tiles: B frags for nt+8 are in flight
        // while nt's 8 WMMAs execute (only 2 waves/SIMD at this LDS size,
        // so rely on ILP rather than TLP to cover L2 latency).
        {
            const __bf16* wl = Woutb + (size_t)layer * kDP * kW;
            const __bf16* bpBase = wl + laneN * kW + koff;

            bf16x16 afr[2][4];
#pragma unroll
            for (int m = 0; m < 2; ++m)
#pragma unroll
                for (int k = 0; k < 4; ++k)
                    afr[m][k] = ld_frag(&Hbf[(m * 16 + laneN) * kW + k * 32 + koff]);

            bf16x16 bcur[4], bnxt[4];
            {
                const __bf16* bp = bpBase + wave * 16 * kW;
#pragma unroll
                for (int k = 0; k < 4; ++k) bcur[k] = ld_frag(bp + k * 32);
            }

            for (int nt = wave; nt < kNT2; nt += 8) {
                int ntn = (nt + 8 < kNT2) ? (nt + 8) : nt;   // clamped prefetch
                {
                    const __bf16* bp = bpBase + ntn * 16 * kW;
#pragma unroll
                    for (int k = 0; k < 4; ++k) bnxt[k] = ld_frag(bp + k * 32);
                }

                int col = nt * 16 + laneN;
                float bias = bout[layer * kDP + col];
#pragma unroll
                for (int m = 0; m < 2; ++m) {
                    v8f acc = {0.f,0.f,0.f,0.f,0.f,0.f,0.f,0.f};
                    acc = wmma_bf16(afr[m][0], bcur[0], acc);
                    acc = wmma_bf16(afr[m][1], bcur[1], acc);
                    acc = wmma_bf16(afr[m][2], bcur[2], acc);
                    acc = wmma_bf16(afr[m][3], bcur[3], acc);
                    int rbase = m * 16 + koff;
#pragma unroll
                    for (int r = 0; r < 8; ++r)
                        Pbuf[(rbase + r) * kPS + col] = acc[r] + bias;
                }
#pragma unroll
                for (int k = 0; k < 4; ++k) bcur[k] = bnxt[k];
            }
        }
        __syncthreads();

        // ---- RQS spline + feature reversal + log-det ----
        {
            int nxt = cur ^ 1;
#pragma unroll 1
            for (int it = 0; it < (kM * kD) / 256; ++it) {
                int idx = it * 256 + t;
                int row = idx >> 6;
                int f   = idx & (kD - 1);
                const float* pp = &Pbuf[row * kPS + f * kP];

                float wv[8], hv[8];
#pragma unroll
                for (int j = 0; j < 8; ++j) wv[j] = pp[j];
#pragma unroll
                for (int j = 0; j < 8; ++j) hv[j] = pp[8 + j];
                float widths[8], heights[8];
                adj_softmax8(wv, widths);
                adj_softmax8(hv, heights);

                float derivs[9];
                derivs[0] = 1.0f; derivs[8] = 1.0f;
#pragma unroll
                for (int j = 0; j < 7; ++j) derivs[j + 1] = softplusf(pp[16 + j]) + kMD;

                float x = Xf[cur][row * kD + f];
                bool inside = (x > -kBI) && (x < kBI);
                float xc = fminf(fmaxf(x, -kBI), kBI);

                // branchless bin select over the cumsum scan
                float xk = -kBI, xk1 = kBI, yk = -kBI, yk1 = kBI, dk = 1.0f, dk1 = 1.0f;
                float xl = -kBI, yl = -kBI;
#pragma unroll
                for (int j = 0; j < 8; ++j) {
                    float xr = xl + widths[j];
                    float yr = yl + heights[j];
                    bool sel = (j == 0) || (xc >= xl);
                    if (sel) { xk = xl; xk1 = xr; yk = yl; yk1 = yr;
                               dk = derivs[j]; dk1 = derivs[j + 1]; }
                    xl = xr; yl = yr;
                }

                float invw = fast_rcp(xk1 - xk);
                float sk   = (yk1 - yk) * invw;
                float xi   = (xc - xk) * invw;
                float omx  = 1.0f - xi;
                float den  = sk + (dk1 + dk - 2.0f * sk) * xi * omx;
                float rden = fast_rcp(den);
                float num  = sk * xi * xi + dk * xi * omx;
                float outv = yk + (yk1 - yk) * num * rden;
                float ldj  = 2.0f * __logf(sk)
                           + __logf(dk1 * xi * xi + 2.0f * sk * xi * omx + dk * omx * omx)
                           - 2.0f * __logf(den);
                outv = inside ? outv : x;
                ldj  = inside ? ldj  : 0.0f;

                Xf[nxt][row * kD + (kD - 1 - f)] = outv;   // x2[:, ::-1]
                atomicAdd(&ldacc[row], ldj);               // ds_add_f32
            }
        }
        __syncthreads();
        cur ^= 1;
    }

    // ---- store outputs ----
#pragma unroll
    for (int i = 0; i < (kM * kD) / 256; ++i) {
        int idx = i * 256 + t;
        out_x[(size_t)row0 * kD + idx] = Xf[cur][idx];
    }
    if (t < kM) out_ld[row0 + t] = ldacc[t];
}

// ---------------- host launcher ----------------
extern "C" void kernel_launch(void* const* d_in, const int* in_sizes, int n_in,
                              void* d_out, int out_size, void* d_ws, size_t ws_size,
                              hipStream_t stream) {
    const float* x     = (const float*)d_in[0];
    const float* y     = (const float*)d_in[1];
    const float* fW1   = (const float*)d_in[2];
    const float* fb1   = (const float*)d_in[3];
    const float* fWout = (const float*)d_in[4];
    const float* fbout = (const float*)d_in[5];
    const float* gW1   = (const float*)d_in[6];
    const float* gb1   = (const float*)d_in[7];
    const float* gWout = (const float*)d_in[8];
    const float* gbout = (const float*)d_in[9];
    float* out = (float*)d_out;

    // workspace layout: masked bf16 weights (~6.3 MB total)
    const size_t nW1  = (size_t)kL * kW * kD;    // 65536
    const size_t nWo  = (size_t)kL * kDP * kW;   // 1507328
    __bf16* fW1b   = (__bf16*)d_ws;
    __bf16* gW1b   = fW1b + nW1;
    __bf16* fWoutb = gW1b + nW1;
    __bf16* gWoutb = fWoutb + nWo;

    cvt_w1_kernel<<<(int)(nW1 / 256), 256, 0, stream>>>(fW1, fW1b);
    cvt_w1_kernel<<<(int)(nW1 / 256), 256, 0, stream>>>(gW1, gW1b);
    cvt_wout_kernel<<<(int)((nWo + 255) / 256), 256, 0, stream>>>(fWout, fWoutb, (int)nWo);
    cvt_wout_kernel<<<(int)((nWo + 255) / 256), 256, 0, stream>>>(gWout, gWoutb, (int)nWo);

    const int blocks = kBATCH / kM;   // 512 workgroups per flow
    // output layout: [xo (16384*64)] [ldf (16384)] [yo (16384*64)] [ldg (16384)]
    float* xo  = out;
    float* ldf = out + (size_t)kBATCH * kD;
    float* yo  = out + (size_t)kBATCH * (kD + 1);
    float* ldg = out + (size_t)kBATCH * (2 * kD + 1);

    maf_flow_kernel<<<blocks, 256, 0, stream>>>(x, fW1b, fb1, fWoutb, fbout, xo, ldf);
    maf_flow_kernel<<<blocks, 256, 0, stream>>>(y, gW1b, gb1, gWoutb, gbout, yo, ldg);
}
